// BitLinear_1855425872140
// MI455X (gfx1250) — 8-bit matrix operand order repaired and checked by reading, compiled, not run on hardware
//
#include <hip/hip_runtime.h>

typedef int v8i __attribute__((ext_vector_type(8)));
typedef int v4i __attribute__((ext_vector_type(4)));
typedef unsigned int u32x4 __attribute__((ext_vector_type(4)));

#define IN_F    4096
#define OUT_F   16384
#define M_TOTAL 8192
#define LDSTR   80

__global__ void k_quant_x(const float* __restrict__ x,
                          signed char* __restrict__ q,
                          float* __restrict__ xscale) {
  const int row = blockIdx.x;
  const float* xr = x + (size_t)row * IN_F;
  __shared__ float red[256];
  float m = 0.f;
  for (int i = threadIdx.x * 4; i < IN_F; i += 1024) {
    float4 v = *(const float4*)(xr + i);
    m = fmaxf(m, fmaxf(fmaxf(fabsf(v.x), fabsf(v.y)), fmaxf(fabsf(v.z), fabsf(v.w))));
  }
  red[threadIdx.x] = m;
  __syncthreads();
  for (int s = 128; s > 0; s >>= 1) {
    if (threadIdx.x < s) red[threadIdx.x] = fmaxf(red[threadIdx.x], red[threadIdx.x + s]);
    __syncthreads();
  }
  const float ma = fmaxf(red[0], 1e-6f);
  if (threadIdx.x == 0) xscale[row] = ma * (1.f / 7.f);
  const float inv = 7.f / ma;
  int* qi = (int*)(q + (size_t)row * IN_F);
  for (int i = threadIdx.x * 4; i < IN_F; i += 1024) {
    float4 v = *(const float4*)(xr + i);
    int b0 = (int)fminf(fmaxf(rintf(v.x * inv), -8.f), 7.f);
    int b1 = (int)fminf(fmaxf(rintf(v.y * inv), -8.f), 7.f);
    int b2 = (int)fminf(fmaxf(rintf(v.z * inv), -8.f), 7.f);
    int b3 = (int)fminf(fmaxf(rintf(v.w * inv), -8.f), 7.f);
    qi[i >> 2] = (b0 & 255) | ((b1 & 255) << 8) | ((b2 & 255) << 16) | ((b3 & 255) << 24);
  }
}

__global__ void k_wstats(const float* __restrict__ w,
                         float* __restrict__ alpha,
                         float* __restrict__ rowsum) {
  const int row = blockIdx.x;
  const float* wr = w + (size_t)row * IN_F;
  __shared__ float red[256];
  float s = 0.f;
  for (int i = threadIdx.x * 4; i < IN_F; i += 1024) {
    float4 v = *(const float4*)(wr + i);
    s += fabsf(v.x) + fabsf(v.y) + fabsf(v.z) + fabsf(v.w);
  }
  red[threadIdx.x] = s;
  __syncthreads();
  for (int st = 128; st > 0; st >>= 1) {
    if (threadIdx.x < st) red[threadIdx.x] += red[threadIdx.x + st];
    __syncthreads();
  }
  if (threadIdx.x == 0) {
    rowsum[row] = red[0];
    alpha[row]  = red[0] * (1.f / (float)IN_F);
  }
}

__global__ void k_thresh(const float* __restrict__ rowsum, float* __restrict__ thr) {
  __shared__ float red[256];
  float s = 0.f;
  for (int i = threadIdx.x; i < OUT_F; i += 256) s += rowsum[i];
  red[threadIdx.x] = s;
  __syncthreads();
  for (int st = 128; st > 0; st >>= 1) {
    if (threadIdx.x < st) red[threadIdx.x] += red[threadIdx.x + st];
    __syncthreads();
  }
  if (threadIdx.x == 0) *thr = 0.05f * red[0] * (1.f / ((float)OUT_F * (float)IN_F));
}

__global__ void k_ternary(const float* __restrict__ w,
                          const float* __restrict__ thr_p,
                          signed char* __restrict__ t8) {
  const float t = *thr_p;
  const size_t i = ((size_t)blockIdx.x * 256 + threadIdx.x) * 4;
  float4 v = *(const float4*)(w + i);
  int b0 = (fabsf(v.x) < t) ? 0 : (v.x > 0.f ? 1 : -1);
  int b1 = (fabsf(v.y) < t) ? 0 : (v.y > 0.f ? 1 : -1);
  int b2 = (fabsf(v.z) < t) ? 0 : (v.z > 0.f ? 1 : -1);
  int b3 = (fabsf(v.w) < t) ? 0 : (v.w > 0.f ? 1 : -1);
  *(int*)(t8 + i) = (b0 & 255) | ((b1 & 255) << 8) | ((b2 & 255) << 16) | ((b3 & 255) << 24);
}

__device__ __forceinline__ void tdm_load_tile(const signed char* gptr, unsigned lds_off) {
  const unsigned long long ga = (unsigned long long)(uintptr_t)gptr;
  u32x4 g0;
  g0.x = 1u;
  g0.y = lds_off;
  g0.z = (unsigned)(ga & 0xFFFFFFFFu);
  g0.w = (unsigned)((ga >> 32) & 0x01FFFFFFu)
         | 0x80000000u;
  v8i g1;
  g1[0] = (1 << 20) | (3 << 22) | (3 << 25);
  g1[1] = (int)((IN_F & 0xFFFF) << 16);
  g1[2] = (int)((IN_F >> 16) | (128 << 16));
  g1[3] = (int)(64 << 16);
  g1[4] = 128;
  g1[5] = IN_F;
  g1[6] = 0;
  g1[7] = 0;
  const v4i z4 = {};
#if defined(__clang_major__) && (__clang_major__ >= 23)
  const v8i z8 = {};
  __builtin_amdgcn_tensor_load_to_lds(g0, g1, z4, z4, z8, 0);
#else
  __builtin_amdgcn_tensor_load_to_lds(g0, g1, z4, z4, 0);
#endif
}

__launch_bounds__(256, 2)
__global__ void k_gemm(const signed char* __restrict__ A,
                       const signed char* __restrict__ B,
                       const float* __restrict__ xscale,
                       const float* __restrict__ alpha,
                       const float* __restrict__ bias,
                       float* __restrict__ out) {
  __shared__ __align__(16) signed char la[2][128 * LDSTR];
  __shared__ __align__(16) signed char lb[2][128 * LDSTR];

  const int tid  = threadIdx.x;
  const int lane = tid & 31;
  const int wave = tid >> 5;
  const int wn = wave & 3;
  const int wm = wave >> 2;
  const int bm = blockIdx.y * 128;
  const int bn = blockIdx.x * 128;
  const int NK = IN_F / 64;

  const unsigned la_o0 = (unsigned)(uintptr_t)&la[0][0];
  const unsigned la_o1 = (unsigned)(uintptr_t)&la[1][0];
  const unsigned lb_o0 = (unsigned)(uintptr_t)&lb[0][0];
  const unsigned lb_o1 = (unsigned)(uintptr_t)&lb[1][0];
  const signed char* gA = A + (size_t)bm * IN_F;
  const signed char* gB = B + (size_t)bn * IN_F;

  if (wave == 0) {
    tdm_load_tile(gA + 0,  la_o0);
    tdm_load_tile(gB + 0,  lb_o0);
    tdm_load_tile(gA + 64, la_o1);
    tdm_load_tile(gB + 64, lb_o1);
  }

  v8i acc[4][2] = {};

  const int aoff = (lane < 16) ? 0 : 16;
  const int boff = (lane < 16) ? 0 : 16;
  const int arow = wm * 64 + (lane & 15);
  const int brow = wn * 32 + (lane & 15);

#pragma unroll 1
  for (int kt = 0; kt < NK; ++kt) {
    const int buf = kt & 1;
    const signed char* lap = buf ? &la[1][0] : &la[0][0];
    const signed char* lbp = buf ? &lb[1][0] : &lb[0][0];
    if (wave == 0) {
      if (kt + 1 < NK) __builtin_amdgcn_s_wait_tensorcnt(2);
      else             __builtin_amdgcn_s_wait_tensorcnt(0);
    }
    __syncthreads();

    union { v4i q[2]; v8i v; } fb[2];
#pragma unroll
    for (int nt = 0; nt < 2; ++nt) {
      const signed char* p = lbp + (size_t)(brow + nt * 16) * LDSTR + boff;
      fb[nt].q[0] = *(const v4i*)(p);
      fb[nt].q[1] = *(const v4i*)(p + 32);
    }
#pragma unroll
    for (int mt = 0; mt < 4; ++mt) {
      union { unsigned long long d[4]; v8i v; } fa;
      const signed char* p = lap + (size_t)(arow + mt * 16) * LDSTR + aoff;
      fa.d[0] = *(const unsigned long long*)(p);
      fa.d[1] = *(const unsigned long long*)(p + 8);
      fa.d[2] = *(const unsigned long long*)(p + 32);
      fa.d[3] = *(const unsigned long long*)(p + 40);
#pragma unroll
      for (int nt = 0; nt < 2; ++nt) {
        acc[mt][nt] = __builtin_amdgcn_wmma_i32_16x16x64_iu8(
            true, fa.v, true, fb[nt].v, acc[mt][nt], false, false);
      }
    }

    __syncthreads();
    if (wave == 0 && kt + 2 < NK) {
      tdm_load_tile(gA + (kt + 2) * 64, buf ? la_o1 : la_o0);
      tdm_load_tile(gB + (kt + 2) * 64, buf ? lb_o1 : lb_o0);
    }
  }

#pragma unroll
  for (int nt = 0; nt < 2; ++nt) {
    const int o  = bn + wn * 32 + nt * 16 + (lane & 15);
    const float av = alpha[o];
    const float bv = bias[o];
#pragma unroll
    for (int mt = 0; mt < 4; ++mt) {
#pragma unroll
      for (int v = 0; v < 8; ++v) {
        const int m = bm + wm * 64 + mt * 16 + v + ((lane >> 4) << 3);
        out[(size_t)m * OUT_F + o] = xscale[m] * av * (float)acc[mt][nt][v] + bv;
      }
    }
  }
}

extern "C" void kernel_launch(void* const* d_in, const int* in_sizes, int n_in,
                              void* d_out, int out_size, void* d_ws, size_t ws_size,
                              hipStream_t stream) {
  (void)in_sizes; (void)n_in; (void)out_size; (void)ws_size;
  const float* x    = (const float*)d_in[0];
  const float* w    = (const float*)d_in[1];
  const float* bias = (const float*)d_in[2];
  float* out = (float*)d_out;

  char* ws = (char*)d_ws;
  signed char* q   = (signed char*)ws;
  signed char* t8  = (signed char*)(ws + ((size_t)32 << 20));
  float* xscale    = (float*)(ws + ((size_t)96 << 20));
  float* alpha     = (float*)(ws + ((size_t)96 << 20) + (32 << 10));
  float* rowsum    = (float*)(ws + ((size_t)96 << 20) + (96 << 10));
  float* thr       = (float*)(ws + ((size_t)96 << 20) + (160 << 10));

  k_quant_x<<<dim3(M_TOTAL), dim3(256), 0, stream>>>(x, q, xscale);
  k_wstats <<<dim3(OUT_F),   dim3(256), 0, stream>>>(w, alpha, rowsum);
  k_thresh <<<dim3(1),       dim3(256), 0, stream>>>(rowsum, thr);
  k_ternary<<<dim3(((size_t)OUT_F * IN_F) / 1024), dim3(256), 0, stream>>>(w, thr, t8);
  k_gemm   <<<dim3(OUT_F / 128, M_TOTAL / 128), dim3(256), 0, stream>>>(
      q, t8, xscale, alpha, bias, out);
}
